// MultiheadSelfAttention_5918464934428
// MI455X (gfx1250) — matrix operand order repaired and checked by reading, not compiled, not run on hardware
//
#include <hip/hip_runtime.h>
#include <hip/hip_bf16.h>

typedef unsigned short u16;
typedef unsigned int   u32;
typedef unsigned long long u64;
typedef __attribute__((ext_vector_type(16))) __bf16 v16bf;
typedef __attribute__((ext_vector_type(8)))  float  v8f;
typedef __attribute__((ext_vector_type(8)))  u16    v8us;
typedef __attribute__((ext_vector_type(16))) u16    v16us;
typedef __attribute__((ext_vector_type(4)))  float  v4f;
typedef __attribute__((ext_vector_type(4)))  int    v4i;
typedef __attribute__((ext_vector_type(8)))  int    v8i;
typedef __attribute__((ext_vector_type(4)))  u32    v4u;

#if defined(__has_builtin)
#if __has_builtin(__builtin_amdgcn_tensor_load_to_lds) && __has_builtin(__builtin_amdgcn_s_wait_tensorcnt)
#define HAVE_TDM 1
#else
#define HAVE_TDM 0
#endif
#else
#define HAVE_TDM 0
#endif

static __device__ __forceinline__ u16 f2bf(float f) {
  union { float f; u32 u; } x; x.f = f;
  u32 u = x.u;
  u += 0x7FFFu + ((u >> 16) & 1u);   // round-to-nearest-even
  return (u16)(u >> 16);
}

static __device__ __forceinline__ v16bf mkfrag(v8us lo, v8us hi) {  // A-fragment (split K chunks)
  v16us u = __builtin_shufflevector(lo, hi, 0,1,2,3,4,5,6,7,8,9,10,11,12,13,14,15);
  return __builtin_bit_cast(v16bf, u);
}

static __device__ __forceinline__ v16bf bfrag(const u16* p) {       // B-fragment (32B contiguous)
  return __builtin_bit_cast(v16bf, *(const v16us*)p);
}

static __device__ __forceinline__ v8f wmma_bf16(v16bf a, v16bf b, v8f c) {
  // (neg_a, A, neg_b, B, c_mod, C, reuse_a, reuse_b)
  return __builtin_amdgcn_wmma_f32_16x16x32_bf16(false, a, false, b, (short)0, c, false, false);
}

// 16-byte memory -> LDS copy; async (no VGPR bounce, ASYNCcnt) when available.
static __device__ __forceinline__ void copy16_to_lds(const u16* g, u16* l) {
#if defined(__has_builtin) && __has_builtin(__builtin_amdgcn_global_load_async_to_lds_b128)
  __builtin_amdgcn_global_load_async_to_lds_b128(
      (__attribute__((address_space(1))) v4i*)(u16*)g,
      (__attribute__((address_space(3))) v4i*)l, 0, 0);
#else
  *(v8us*)l = *(const v8us*)g;
#endif
}

static __device__ __forceinline__ void wait_lds_copies() {
#if defined(__has_builtin) && __has_builtin(__builtin_amdgcn_global_load_async_to_lds_b128) && \
    __has_builtin(__builtin_amdgcn_s_wait_asynccnt)
  __builtin_amdgcn_s_wait_asynccnt(0);
#endif
}

static __device__ __forceinline__ u32 lds_off(const u16* p) {       // LDS byte address for D#
  return (u32)(size_t)(__attribute__((address_space(3))) const u16*)p;
}

// TDM: one descriptor moves a tile_d1 x tile_d0 (elements, 2B each) tile from a
// row-major tensor (row stride stride0 elements) into LDS, padding +8 DWORDs
// after every 16 DWORDs -> 48-element LDS row stride (matches fragment layout).
static __device__ __forceinline__ void tdm_load_tile(const u16* gsrc, u32 lds_byte,
                                                     u32 tensor_d0, u32 tensor_d1,
                                                     u32 tile_d0, u32 tile_d1, u32 stride0) {
#if HAVE_TDM
  u64 ga = (u64)(size_t)gsrc;
  v4u g0;
  g0[0] = 1u;                                   // count=1 (valid), user mode, no gather
  g0[1] = lds_byte;                             // lds_addr
  g0[2] = (u32)ga;                              // global_addr[31:0]
  g0[3] = (u32)(ga >> 32) | (2u << 30);         // global_addr[56:32] | type=2 ("image")
  v8i g1;
  g1[0] = (int)((1u << 16)                      // data_size = 2 bytes
              | (1u << 20)                      // pad_enable
              | (3u << 22)                      // pad_interval: 16 DWORDs
              | (7u << 25));                    // pad_amount : 8 DWORDs
  g1[1] = (int)((tensor_d0 & 0xFFFFu) << 16);   // atomic_barrier_addr=0 | tensor_dim0[15:0]
  g1[2] = (int)((tensor_d0 >> 16) | ((tensor_d1 & 0xFFFFu) << 16));
  g1[3] = (int)((tensor_d1 >> 16) | (tile_d0 << 16));
  g1[4] = (int)(tile_d1 & 0xFFFFu);             // tile_dim1 | tile_dim2=0
  g1[5] = (int)stride0;                         // tensor_dim0_stride[31:0]
  g1[6] = 0;                                    // stride0[47:32] | stride1[15:0]
  g1[7] = 0;
  v4i z4 = {0, 0, 0, 0};                        // 2-D tensor: groups 2/3 unused
  v8i z8 = {0, 0, 0, 0, 0, 0, 0, 0};
  __builtin_amdgcn_tensor_load_to_lds(g0, g1, z4, z4, z8, 0);
#else
  (void)gsrc; (void)lds_byte; (void)tensor_d0; (void)tensor_d1;
  (void)tile_d0; (void)tile_d1; (void)stride0;
#endif
}

static __device__ __forceinline__ void wait_tdm() {
#if HAVE_TDM
  __builtin_amdgcn_s_wait_tensorcnt(0);
#endif
}

// ---------------------------------------------------------------- convert ---
__global__ __launch_bounds__(256) void cvt_f32_bf16(const float* __restrict__ src,
                                                    u16* __restrict__ dst, int n) {
  int i = (blockIdx.x * 256 + threadIdx.x) * 8;
  if (i + 8 <= n) {
    v4f a = *(const v4f*)(src + i);
    v4f b = *(const v4f*)(src + i + 4);
    v8us o;
#pragma unroll
    for (int j = 0; j < 4; ++j) { o[j] = f2bf(a[j]); o[4 + j] = f2bf(b[j]); }
    *(v8us*)(dst + i) = o;
  }
}

// ------------------------------------------------------------------- GEMM ---
// Y[m,n] = scale * sum_k X[m,k] * W[n,k].   M=4096, N=K=1024.
// QKV=true : store bf16 to out[b][h][s][dh] (m=b*2048+s, n=h*64+dh)
// QKV=false: store f32  to out[m*1024+n]
// Double-buffered LDS; tiles staged by the Tensor Data Mover (TENSORcnt) when
// available, else per-lane async-to-LDS copies (ASYNCcnt).
template<bool QKV>
__global__ __launch_bounds__(256) void gemm_xwt(const u16* __restrict__ X,
                                                const u16* __restrict__ W,
                                                void* __restrict__ outp, float scale) {
  __shared__ u16 sA[2][128 * 48];   // 128 rows x 32 k (+pad to 48) bf16, 96B row stride
  __shared__ u16 sB[2][128 * 48];
  const int tid = threadIdx.x, lane = tid & 31, wv = tid >> 5;
  const int wm = wv & 1, wn = wv >> 1;            // 2x4 wave grid
  const int r = lane & 15, hif = lane >> 4;
  const int kh = hif * 8, kb = hif * 16;
  const int bm = blockIdx.y * 128, bn = blockIdx.x * 128;

  v8f acc[4][2] = {};

  auto stage = [&](int buf, int k0) {
    if (HAVE_TDM) {
      if (wv == 0) {                              // one wave drives the TDM
        tdm_load_tile(&X[(size_t)bm * 1024 + k0], lds_off(&sA[buf][0]), 1024, 4096, 32, 128, 1024);
        tdm_load_tile(&W[(size_t)bn * 1024 + k0], lds_off(&sB[buf][0]), 1024, 1024, 32, 128, 1024);
      }
    } else {
#pragma unroll
      for (int i = 0; i < 2; ++i) {
        int slot = tid + i * 256, row = slot >> 2, ch = slot & 3;
        copy16_to_lds(&X[(size_t)(bm + row) * 1024 + k0 + ch * 8], &sA[buf][row * 48 + ch * 8]);
        copy16_to_lds(&W[(size_t)(bn + row) * 1024 + k0 + ch * 8], &sB[buf][row * 48 + ch * 8]);
      }
    }
  };
  auto stage_wait = [&]() {
    if (HAVE_TDM) { if (wv == 0) wait_tdm(); }
    else wait_lds_copies();
  };

  stage(0, 0);
  stage_wait();
  __syncthreads();

  for (int k0 = 0; k0 < 1024; k0 += 32) {
    const int cur = (k0 >> 5) & 1;
    if (k0 + 32 < 1024) {
      stage(cur ^ 1, k0 + 32);                    // overlap next-tile DMA with compute
      __builtin_prefetch(&X[(size_t)(bm + (tid >> 1)) * 1024 + k0 + 64 + (tid & 1) * 16], 0, 1);
      __builtin_prefetch(&W[(size_t)(bn + (tid >> 1)) * 1024 + k0 + 64 + (tid & 1) * 16], 0, 1);
    }

    v16bf af[4], bfr[2];
#pragma unroll
    for (int mt = 0; mt < 4; ++mt) {              // A-fragment: contiguous K16 per half
      const u16* p = &sA[cur][(wm * 64 + mt * 16 + r) * 48];
      af[mt] = mkfrag(*(const v8us*)&p[kb], *(const v8us*)&p[kb + 8]);
    }
#pragma unroll
    for (int nt = 0; nt < 2; ++nt)                // B-fragment: contiguous K16 per half
      bfr[nt] = bfrag(&sB[cur][(wn * 32 + nt * 16 + r) * 48 + kb]);
#pragma unroll
    for (int mt = 0; mt < 4; ++mt)
#pragma unroll
      for (int nt = 0; nt < 2; ++nt)
        acc[mt][nt] = wmma_bf16(af[mt], bfr[nt], acc[mt][nt]);

    stage_wait();
    __syncthreads();
  }

#pragma unroll
  for (int mt = 0; mt < 4; ++mt)
#pragma unroll
    for (int nt = 0; nt < 2; ++nt)
#pragma unroll
      for (int e = 0; e < 8; ++e) {
        int m = bm + wm * 64 + mt * 16 + hif * 8 + e;
        int n = bn + wn * 32 + nt * 16 + r;
        float v = acc[mt][nt][e] * scale;
        if (QKV) {
          int b = m >> 11, s = m & 2047, h = n >> 6, d = n & 63;
          ((u16*)outp)[(((size_t)b * 16 + h) * 2048 + s) * 64 + d] = f2bf(v);
        } else {
          ((float*)outp)[(size_t)m * 1024 + n] = v;
        }
      }
}

// -------------------------------------------------------- flash attention ---
// Q,K,V: bf16 [b][h][2048][64] (Q pre-scaled by 1/8). O: bf16 [b][s][1024].
__global__ __launch_bounds__(128) void flash_attn(const u16* __restrict__ Q,
                                                  const u16* __restrict__ K,
                                                  const u16* __restrict__ V,
                                                  u16* __restrict__ O) {
  __shared__ u16 sK[64 * 80];    // [kpos][dh]   (row stride 160B, 32B aligned frags)
  __shared__ u16 sVt[64 * 80];   // [dh][kpos]   (V transposed)
  __shared__ u16 sP[64 * 80];    // [qrow][kpos] per-wave strips
  const int tid = threadIdx.x, lane = tid & 31, wv = tid >> 5;
  const int r = lane & 15, hif = lane >> 4;
  const int kh = hif * 8, kb = hif * 16;
  const int qt = blockIdx.x, bh = blockIdx.y;
  const size_t base = (size_t)bh * 2048 * 64;

  const u16* qp = Q + base + (size_t)(qt * 64 + wv * 16 + r) * 64;
  v16bf qf[2];
  qf[0] = mkfrag(*(const v8us*)&qp[kb],      *(const v8us*)&qp[kb + 8]);
  qf[1] = mkfrag(*(const v8us*)&qp[32 + kb], *(const v8us*)&qp[32 + kb + 8]);

  v8f acc[4] = {};
  float mr[8], lr[8];
#pragma unroll
  for (int e = 0; e < 8; ++e) { mr[e] = -__builtin_inff(); lr[e] = 0.f; }

  for (int kt = 0; kt <= qt; ++kt) {
#pragma unroll
    for (int i = 0; i < 4; ++i) {               // cooperative K/V staging
      int slot = tid + i * 128, row = slot >> 3, ch = slot & 7;
      copy16_to_lds(&K[base + (size_t)(kt * 64 + row) * 64 + ch * 8], &sK[row * 80 + ch * 8]);
      v8us vvv = *(const v8us*)&V[base + (size_t)(kt * 64 + row) * 64 + ch * 8];
#pragma unroll
      for (int j = 0; j < 8; ++j) sVt[(ch * 8 + j) * 80 + row] = vvv[j];   // V^T
    }
    wait_lds_copies();
    __syncthreads();

    v8f s[4];
#pragma unroll
    for (int nt = 0; nt < 4; ++nt) {            // S = Q K^T  (B^T = K rows)
      const u16* kp = &sK[(nt * 16 + r) * 80];
      v16bf b0 = bfrag(&kp[kb]);
      v16bf b1 = bfrag(&kp[32 + kb]);
      v8f c = {};
      c = wmma_bf16(qf[0], b0, c);
      c = wmma_bf16(qf[1], b1, c);
      s[nt] = c;
    }
    if (kt == qt) {                             // causal mask on diagonal tile
#pragma unroll
      for (int nt = 0; nt < 4; ++nt)
#pragma unroll
        for (int e = 0; e < 8; ++e)
          if (nt * 16 + r > wv * 16 + hif * 8 + e) s[nt][e] = -__builtin_inff();
    }
#pragma unroll
    for (int e = 0; e < 8; ++e) {               // online softmax (16-lane rows)
      float rm = fmaxf(fmaxf(s[0][e], s[1][e]), fmaxf(s[2][e], s[3][e]));
      rm = fmaxf(rm, __shfl_xor(rm, 1, 32));
      rm = fmaxf(rm, __shfl_xor(rm, 2, 32));
      rm = fmaxf(rm, __shfl_xor(rm, 4, 32));
      rm = fmaxf(rm, __shfl_xor(rm, 8, 32));
      float mn = fmaxf(mr[e], rm);
      float al = __expf(mr[e] - mn);
      float rs = 0.f;
#pragma unroll
      for (int nt = 0; nt < 4; ++nt) { float p = __expf(s[nt][e] - mn); s[nt][e] = p; rs += p; }
      rs += __shfl_xor(rs, 1, 32);
      rs += __shfl_xor(rs, 2, 32);
      rs += __shfl_xor(rs, 4, 32);
      rs += __shfl_xor(rs, 8, 32);
      lr[e] = lr[e] * al + rs;
      mr[e] = mn;
#pragma unroll
      for (int ot = 0; ot < 4; ++ot) acc[ot][e] *= al;
    }
    u16* pp = &sP[wv * 16 * 80];                // wave-private P strip
#pragma unroll
    for (int nt = 0; nt < 4; ++nt)
#pragma unroll
      for (int e = 0; e < 8; ++e)
        pp[(hif * 8 + e) * 80 + nt * 16 + r] = f2bf(s[nt][e]);

    const u16* pr = &pp[r * 80];                // reload P in A-fragment layout
    v16bf pf0 = mkfrag(*(const v8us*)&pr[kb],      *(const v8us*)&pr[kb + 8]);
    v16bf pf1 = mkfrag(*(const v8us*)&pr[32 + kb], *(const v8us*)&pr[32 + kb + 8]);
#pragma unroll
    for (int ot = 0; ot < 4; ++ot) {            // O += P V  (B^T = V^T rows)
      const u16* vp = &sVt[(ot * 16 + r) * 80];
      v16bf v0 = bfrag(&vp[kb]);
      v16bf v1 = bfrag(&vp[32 + kb]);
      acc[ot] = wmma_bf16(pf0, v0, acc[ot]);
      acc[ot] = wmma_bf16(pf1, v1, acc[ot]);
    }
    __syncthreads();
  }

  const int b = bh >> 4, h = bh & 15;
#pragma unroll
  for (int e = 0; e < 8; ++e) {
    float inv = 1.0f / lr[e];
    int srow = qt * 64 + wv * 16 + hif * 8 + e;
#pragma unroll
    for (int ot = 0; ot < 4; ++ot) {
      int d = h * 64 + ot * 16 + r;
      O[((size_t)b * 2048 + srow) * 1024 + d] = f2bf(acc[ot][e] * inv);
    }
  }
}

// ----------------------------------------------------------------- launch ---
extern "C" void kernel_launch(void* const* d_in, const int* in_sizes, int n_in,
                              void* d_out, int out_size, void* d_ws, size_t ws_size,
                              hipStream_t stream) {
  (void)in_sizes; (void)n_in; (void)out_size; (void)ws_size;
  const float* x  = (const float*)d_in[0];
  const float* Wq = (const float*)d_in[1];
  const float* Wk = (const float*)d_in[2];
  const float* Wv = (const float*)d_in[3];
  const float* Wo = (const float*)d_in[4];

  u16* ws = (u16*)d_ws;
  size_t o = 0;
  u16* xb  = ws + o; o += (size_t)4096 * 1024;   // x  bf16
  u16* wqb = ws + o; o += (size_t)1024 * 1024;
  u16* wkb = ws + o; o += (size_t)1024 * 1024;
  u16* wvb = ws + o; o += (size_t)1024 * 1024;
  u16* wob = ws + o; o += (size_t)1024 * 1024;
  u16* Qb  = ws + o; o += (size_t)4096 * 1024;   // [b][h][s][dh]
  u16* Kb  = ws + o; o += (size_t)4096 * 1024;
  u16* Vb  = ws + o; o += (size_t)4096 * 1024;
  u16* ao  = ws + o; o += (size_t)4096 * 1024;   // attn out [b][s][d]

  cvt_f32_bf16<<<2048, 256, 0, stream>>>(x,  xb,  4194304);
  cvt_f32_bf16<<<512,  256, 0, stream>>>(Wq, wqb, 1048576);
  cvt_f32_bf16<<<512,  256, 0, stream>>>(Wk, wkb, 1048576);
  cvt_f32_bf16<<<512,  256, 0, stream>>>(Wv, wvb, 1048576);
  cvt_f32_bf16<<<512,  256, 0, stream>>>(Wo, wob, 1048576);

  dim3 g(8, 32);  // N/128, M/128
  gemm_xwt<true ><<<g, 256, 0, stream>>>(xb, wqb, Qb, 0.125f);  // scale folded into Q
  gemm_xwt<true ><<<g, 256, 0, stream>>>(xb, wkb, Kb, 1.0f);
  gemm_xwt<true ><<<g, 256, 0, stream>>>(xb, wvb, Vb, 1.0f);

  flash_attn<<<dim3(32, 32), 128, 0, stream>>>(Qb, Kb, Vb, ao);

  gemm_xwt<false><<<g, 256, 0, stream>>>(ao, wob, (float*)d_out, 1.0f);
}
